// MultiHeadAttentionBlock_22711787061337
// MI455X (gfx1250) — matrix operand order repaired and checked by reading, not compiled, not run on hardware
//
#include <hip/hip_runtime.h>
#include <hip/hip_bf16.h>

// MI455X / gfx1250, wave32. MHA block: B=2, S=2048, D=1024, H=16, DK=64.
// bf16 WMMA (v_wmma_f32_16x16x32_bf16) everywhere; fp32 accumulate.
// GEMM A-operand staged to LDS with the Tensor Data Mover (tensor_load_to_lds,
// TENSORcnt); attention is a per-wave flash loop (no block barriers).

typedef __bf16 bf16_t;
typedef __attribute__((ext_vector_type(16))) __bf16 v16bf;
typedef __attribute__((ext_vector_type(8)))  __bf16 v8bf;
typedef __attribute__((ext_vector_type(8)))  float  v8f;
typedef __attribute__((ext_vector_type(4)))  unsigned int v4u;
typedef __attribute__((ext_vector_type(8)))  int v8i;
typedef __attribute__((ext_vector_type(4)))  int v4i;

#define B_  2
#define S_  2048
#define D_  1024
#define H_  16
#define DK_ 64

__device__ __forceinline__ v8f wmma_bf16f32(v16bf a, v16bf b, v8f c) {
  // (neg_a, A, neg_b, B, c_mod, C, reuse_a, reuse_b)
  return __builtin_amdgcn_wmma_f32_16x16x32_bf16(false, a, false, b, (short)0, c,
                                                 false, false);
}

// ---- fragment loaders (CDNA5 wave32 VGPR layouts, ISA 05_wmma §7.12.2) ----
// A (16x32 bf16): lane<16 holds row M=lane, K = {kb..kb+7, kb+16..kb+23};
//                 lane>=16 holds row M=lane-16, K = {kb+8..kb+15, kb+24..kb+31}.
__device__ __forceinline__ v16bf load_a_frag(const bf16_t* base, size_t row,
                                             int ld, int kbase, int lane) {
  const bf16_t* p = base + row * (size_t)ld + (size_t)(kbase + ((lane >> 4) << 3));
  v8bf lo = *(const v8bf*)p;
  v8bf hi = *(const v8bf*)(p + 16);
  v16bf r;
#pragma unroll
  for (int i = 0; i < 8; ++i) { r[i] = lo[i]; r[i + 8] = hi[i]; }
  return r;
}

// B (32x16 bf16) where memory holds B^T row-major (row n, contiguous k):
// lane holds column N = lane&15, K = kb + (lane>=16?8:0) + {0..7, 16..23}.
__device__ __forceinline__ v16bf load_b_frag(const bf16_t* base, size_t nrow,
                                             int ld, int kbase, int lane) {
  const bf16_t* p = base + nrow * (size_t)ld + (size_t)(kbase + ((lane >> 4) << 3));
  v8bf lo = *(const v8bf*)p;
  v8bf hi = *(const v8bf*)(p + 16);
  v16bf r;
#pragma unroll
  for (int i = 0; i < 8; ++i) { r[i] = lo[i]; r[i + 8] = hi[i]; }
  return r;
}

// ---- Tensor Data Mover: 2D bf16 tile (rows x cols) global -> LDS ----
// D# layout per CDNA5 ISA (group0 128b, group1 256b).
__device__ __forceinline__ void tdm_load_2d_bf16(const void* gaddr,
                                                 unsigned lds_off,
                                                 unsigned rows, unsigned cols,
                                                 unsigned row_stride_elems) {
  unsigned long long ga = (unsigned long long)(size_t)gaddr;
  v4u g0;
  g0[0] = 1u;                                        // count=1 (user mode)
  g0[1] = lds_off;                                   // lds_addr (bytes)
  g0[2] = (unsigned)(ga & 0xffffffffu);              // global_addr[31:0]
  g0[3] = ((unsigned)(ga >> 32) & 0x01ffffffu)       // global_addr[56:32]
          | (2u << 30);                              // type=2 ("image")
  v8i g1;
  g1[0] = (int)(1u << 16);                           // data_size=1 (2B); wg_mask=0
  g1[1] = (int)((cols & 0xffffu) << 16);             // tensor_dim0[15:0]
  g1[2] = (int)(((cols >> 16) & 0xffffu)             // tensor_dim0[31:16]
                | ((rows & 0xffffu) << 16));         // tensor_dim1[15:0]
  g1[3] = (int)(((rows >> 16) & 0xffffu)             // tensor_dim1[31:16]
                | ((cols & 0xffffu) << 16));         // tile_dim0
  g1[4] = (int)(rows & 0xffffu);                     // tile_dim1 (tile_dim2=0)
  g1[5] = (int)row_stride_elems;                     // tensor_dim0_stride[31:0]
  g1[6] = 0;                                         // stride hi / dim1_stride lo
  g1[7] = 0;
  v4i z = {0, 0, 0, 0};
#if __clang_major__ >= 23
  v8i z8 = {0, 0, 0, 0, 0, 0, 0, 0};
  __builtin_amdgcn_tensor_load_to_lds(g0, g1, z, z, z8, 0);
#else
  __builtin_amdgcn_tensor_load_to_lds(g0, g1, z, z, 0);
#endif
}

// ---- fp32 -> bf16 convert ----
__global__ __launch_bounds__(256) void cvt_kernel(const float* __restrict__ in,
                                                  bf16_t* __restrict__ out, int n) {
  for (int i = blockIdx.x * blockDim.x + threadIdx.x; i < n;
       i += gridDim.x * blockDim.x)
    out[i] = (bf16_t)in[i];
}

// ---- GEMM: C[M,1024] = act[M,1024] @ W[1024,1024]^T  (y = x W^T) ----
// Block tile 128(M) x 128(N); 8 waves as 4(M) x 2(N); wave tile 32 x 64.
// A 128x32 k-slice staged to LDS by the TDM (wave 0), B streamed from L2.
// MODE 0: Q proj (+RoPE, head-major bf16 [B,H,S,DK])
// MODE 1: K proj (+RoPE, head-major bf16)
// MODE 2: V proj (transposed bf16 [B,H,DK,S])
// MODE 3: out proj (fp32 [B,S,D])
template <int MODE>
__global__ __launch_bounds__(256) void gemm_kernel(const bf16_t* __restrict__ A,
                                                   const bf16_t* __restrict__ W,
                                                   void* __restrict__ out) {
  __shared__ bf16_t atile[128 * 32];                 // 8KB k-slice of A
  const int lane = threadIdx.x & 31;
  const int wave = threadIdx.x >> 5;                 // 8 waves
  const int mw = (wave >> 1) * 32;                   // 0,32,64,96
  const int nw = (wave & 1) * 64;                    // 0,64
  const int m0 = blockIdx.y * 128;
  const int n0 = blockIdx.x * 128 + nw;

  v8f c[2][4];
#pragma unroll
  for (int i = 0; i < 2; ++i)
#pragma unroll
    for (int t = 0; t < 4; ++t) c[i][t] = v8f{};

  for (int k = 0; k < D_; k += 32) {
    __syncthreads();                                 // prior reads done
    if (wave == 0) {
      tdm_load_2d_bf16(A + (size_t)m0 * D_ + k, /*lds_off=*/0u,
                       /*rows=*/128u, /*cols=*/32u, /*row_stride=*/(unsigned)D_);
      __builtin_amdgcn_s_wait_tensorcnt(0);
    }
    __syncthreads();                                 // tile visible to all waves
    const v16bf a0 = load_a_frag(atile, (size_t)(mw + (lane & 15)), 32, 0, lane);
    const v16bf a1 = load_a_frag(atile, (size_t)(mw + 16 + (lane & 15)), 32, 0, lane);
#pragma unroll
    for (int t = 0; t < 4; ++t) {
      const size_t nrow = (size_t)(n0 + t * 16 + (lane & 15));
      __builtin_prefetch((const void*)(W + nrow * D_ + k + 128), 0, 1);
      v16bf b = load_b_frag(W, nrow, D_, k, lane);
      c[0][t] = wmma_bf16f32(a0, b, c[0][t]);
      c[1][t] = wmma_bf16f32(a1, b, c[1][t]);
    }
  }

  // C layout: c[i][t][r] -> (M = mw + i*16 + r + (lane>=16?8:0), N = t*16 + (lane&15))
  const int hi8 = (lane >> 4) << 3;
#pragma unroll
  for (int i = 0; i < 2; ++i) {
#pragma unroll
    for (int t = 0; t < 4; ++t) {
#pragma unroll
      for (int r = 0; r < 8; ++r) {
        const int gm = m0 + mw + i * 16 + r + hi8;   // global row in [0, B*S)
        const int gn = n0 + t * 16 + (lane & 15);
        float val = c[i][t][r];
        if constexpr (MODE == 3) {
          ((float*)out)[(size_t)gm * D_ + gn] = val;
        } else {
          const int b = gm >> 11, s = gm & (S_ - 1);
          const int h = gn >> 6, dk = gn & (DK_ - 1);
          if constexpr (MODE <= 1) {
            // RoPE, pairwise-interleaved; partner element is in adjacent lane.
            const float partner = __shfl_xor(val, 1, 32);
            const int jj = dk & 31;
            const float ang = (float)s * __powf(10000.0f, -(float)jj * (1.0f / 32.0f));
            const float cv = __cosf(ang), sv = __sinf(ang);
            val = (dk & 1) ? (partner * sv + val * cv) : (val * cv - partner * sv);
          }
          bf16_t* o = (bf16_t*)out;
          if constexpr (MODE == 2)  // V transposed: [B,H,DK,S]
            o[(((size_t)(b * H_ + h) * DK_) + dk) * S_ + s] = (bf16_t)val;
          else                      // Q/K head-major: [B,H,S,DK]
            o[(((size_t)(b * H_ + h) * S_) + s) * DK_ + dk] = (bf16_t)val;
        }
      }
    }
  }
}

// ---- causal flash attention: per wave 16 queries, 32-key blocks ----
__global__ __launch_bounds__(256) void attn_kernel(const bf16_t* __restrict__ Qh,
                                                   const bf16_t* __restrict__ Kh,
                                                   const bf16_t* __restrict__ Vt,
                                                   bf16_t* __restrict__ X) {
  __shared__ bf16_t pbuf[8 * 16 * 32];               // per-wave P patch (1KB each)
  const int lane = threadIdx.x & 31;
  const int wave = threadIdx.x >> 5;
  const int bh = blockIdx.y;                         // b*16 + h
  const int b = bh >> 4, h = bh & 15;
  const int qbase = blockIdx.x * 128 + wave * 16;
  const int hi8 = (lane >> 4) << 3;

  const bf16_t* Qp = Qh + (size_t)bh * S_ * DK_;
  const bf16_t* Kp = Kh + (size_t)bh * S_ * DK_;
  const bf16_t* Vp = Vt + (size_t)bh * DK_ * S_;
  bf16_t* myp = pbuf + wave * (16 * 32);

  const v16bf aq0 = load_a_frag(Qp, (size_t)qbase + (lane & 15), DK_, 0, lane);
  const v16bf aq1 = load_a_frag(Qp, (size_t)qbase + (lane & 15), DK_, 32, lane);

  v8f o[4] = {v8f{}, v8f{}, v8f{}, v8f{}};
  float rm[8], rl[8];
#pragma unroll
  for (int r = 0; r < 8; ++r) { rm[r] = -3.0e38f; rl[r] = 0.0f; }

  const float scale = 0.125f;                        // 1/sqrt(64)
  const int nkb = (qbase + 16 + 31) >> 5;            // wave-uniform trip count

  for (int j = 0; j < nkb; ++j) {
    const int kb = j * 32;
    // S = Q @ K^T  (two 16-col tiles, k over DK)
    v8f s0{}, s1{};
    {
      v16bf bk = load_b_frag(Kp, (size_t)kb + (lane & 15), DK_, 0, lane);
      s0 = wmma_bf16f32(aq0, bk, s0);
      bk = load_b_frag(Kp, (size_t)kb + (lane & 15), DK_, 32, lane);
      s0 = wmma_bf16f32(aq1, bk, s0);
      bk = load_b_frag(Kp, (size_t)kb + 16 + (lane & 15), DK_, 0, lane);
      s1 = wmma_bf16f32(aq0, bk, s1);
      bk = load_b_frag(Kp, (size_t)kb + 16 + (lane & 15), DK_, 32, lane);
      s1 = wmma_bf16f32(aq1, bk, s1);
    }
    const int kn0 = kb + (lane & 15), kn1 = kb + 16 + (lane & 15);
    // scale + causal mask + online softmax (rows live across 16-lane half-group)
#pragma unroll
    for (int r = 0; r < 8; ++r) {
      const int q = qbase + r + hi8;
      float v0 = s0[r] * scale; if (kn0 > q) v0 = -1.0e9f;
      float v1 = s1[r] * scale; if (kn1 > q) v1 = -1.0e9f;
      float mx = fmaxf(v0, v1);
#pragma unroll
      for (int d = 1; d < 16; d <<= 1) mx = fmaxf(mx, __shfl_xor(mx, d, 32));
      const float mnew = fmaxf(rm[r], mx);
      const float corr = __expf(rm[r] - mnew);
      rm[r] = mnew;
      const float p0 = __expf(v0 - mnew), p1 = __expf(v1 - mnew);
      float sum = p0 + p1;
#pragma unroll
      for (int d = 1; d < 16; d <<= 1) sum += __shfl_xor(sum, d, 32);
      rl[r] = rl[r] * corr + sum;
#pragma unroll
      for (int t = 0; t < 4; ++t) o[t][r] *= corr;
      s0[r] = p0; s1[r] = p1;
    }
    // P: C-layout -> A-fragment re-layout through LDS (same-wave RAW, DScnt)
#pragma unroll
    for (int r = 0; r < 8; ++r) {
      const int m = r + hi8;
      myp[m * 32 + (lane & 15)]      = (bf16_t)s0[r];
      myp[m * 32 + 16 + (lane & 15)] = (bf16_t)s1[r];
    }
    v16bf ap;
    {
      const bf16_t* p = myp + (lane & 15) * 32 + ((lane >> 4) << 3);
      v8bf lo = *(const v8bf*)p;
      v8bf hi = *(const v8bf*)(p + 16);
#pragma unroll
      for (int i = 0; i < 8; ++i) { ap[i] = lo[i]; ap[i + 8] = hi[i]; }
    }
    // O += P @ V   (V stored transposed: row = dk, contiguous keys)
#pragma unroll
    for (int t = 0; t < 4; ++t) {
      v16bf bv = load_b_frag(Vp, (size_t)(t * 16 + (lane & 15)), S_, kb, lane);
      o[t] = wmma_bf16f32(ap, bv, o[t]);
    }
  }

  // normalize and scatter to [B,S,D] bf16 (head h occupies cols h*64..h*64+63)
#pragma unroll
  for (int r = 0; r < 8; ++r) {
    const int q = qbase + r + hi8;
    const float inv = 1.0f / rl[r];
#pragma unroll
    for (int t = 0; t < 4; ++t) {
      const int dk = t * 16 + (lane & 15);
      X[((size_t)b * S_ + q) * D_ + h * DK_ + dk] = (bf16_t)(o[t][r] * inv);
    }
  }
}

extern "C" void kernel_launch(void* const* d_in, const int* in_sizes, int n_in,
                              void* d_out, int out_size, void* d_ws, size_t ws_size,
                              hipStream_t stream) {
  const float* q  = (const float*)d_in[0];
  const float* k  = (const float*)d_in[1];
  const float* v  = (const float*)d_in[2];
  // d_in[3] = mask (causal, rebuilt analytically in-kernel)
  const float* wq = (const float*)d_in[4];
  const float* wk = (const float*)d_in[5];
  const float* wv = (const float*)d_in[6];
  const float* wo = (const float*)d_in[7];

  const size_t NACT = (size_t)B_ * S_ * D_;   // 4 Mi elements
  const size_t NW   = (size_t)D_ * D_;        // 1 Mi elements
  char* ws = (char*)d_ws;
  bf16_t* qb  = (bf16_t*)ws; ws += NACT * sizeof(bf16_t);
  bf16_t* kbf = (bf16_t*)ws; ws += NACT * sizeof(bf16_t);
  bf16_t* vb  = (bf16_t*)ws; ws += NACT * sizeof(bf16_t);
  bf16_t* wqb = (bf16_t*)ws; ws += NW * sizeof(bf16_t);
  bf16_t* wkb = (bf16_t*)ws; ws += NW * sizeof(bf16_t);
  bf16_t* wvb = (bf16_t*)ws; ws += NW * sizeof(bf16_t);
  bf16_t* wob = (bf16_t*)ws; ws += NW * sizeof(bf16_t);
  bf16_t* Qh  = (bf16_t*)ws; ws += NACT * sizeof(bf16_t);
  bf16_t* Kh  = (bf16_t*)ws; ws += NACT * sizeof(bf16_t);
  bf16_t* Vt  = (bf16_t*)ws; ws += NACT * sizeof(bf16_t);
  bf16_t* X   = (bf16_t*)ws; ws += NACT * sizeof(bf16_t);
  (void)ws_size; (void)in_sizes; (void)n_in; (void)out_size;

  // 1) fp32 -> bf16
  cvt_kernel<<<2048, 256, 0, stream>>>(q,  qb,  (int)NACT);
  cvt_kernel<<<2048, 256, 0, stream>>>(k,  kbf, (int)NACT);
  cvt_kernel<<<2048, 256, 0, stream>>>(v,  vb,  (int)NACT);
  cvt_kernel<<<1024, 256, 0, stream>>>(wq, wqb, (int)NW);
  cvt_kernel<<<1024, 256, 0, stream>>>(wk, wkb, (int)NW);
  cvt_kernel<<<1024, 256, 0, stream>>>(wv, wvb, (int)NW);
  cvt_kernel<<<1024, 256, 0, stream>>>(wo, wob, (int)NW);

  // 2) projections (+RoPE for Q/K, transpose-store for V)
  dim3 ggrid(D_ / 128, (B_ * S_) / 128);  // 8 x 32
  gemm_kernel<0><<<ggrid, 256, 0, stream>>>(qb,  wqb, (void*)Qh);
  gemm_kernel<1><<<ggrid, 256, 0, stream>>>(kbf, wkb, (void*)Kh);
  gemm_kernel<2><<<ggrid, 256, 0, stream>>>(vb,  wvb, (void*)Vt);

  // 3) causal attention
  dim3 agrid(S_ / 128, B_ * H_);          // 16 x 32
  attn_kernel<<<agrid, 256, 0, stream>>>(Qh, Kh, Vt, X);

  // 4) output projection -> fp32 d_out
  gemm_kernel<3><<<ggrid, 256, 0, stream>>>(X, wob, d_out);
}
